// ConvLTC_19825569039009
// MI455X (gfx1250) — matrix operand order repaired and checked by reading, not compiled, not run on hardware
//
#include <hip/hip_runtime.h>
#include <hip/hip_bf16.h>
#include <math.h>

typedef __attribute__((ext_vector_type(16))) _Float16 v16h;
typedef __attribute__((ext_vector_type(8)))  _Float16 v8h;
typedef __attribute__((ext_vector_type(8)))  float    v8f;

// ---- problem constants (match reference) ----
static constexpr int kB = 4, kT = 32, kH = 160, kW = 320;
static constexpr int kNF = 8, kFPF = 8, kSDIM = 8;
static constexpr int kU = 64, kS = kNF * kFPF + kSDIM; // 72
static constexpr int kBT = kB * kT;                    // 128
static constexpr float kEPS = 1e-8f;

__device__ __forceinline__ float sigm_(float x) { return 1.0f / (1.0f + __expf(-x)); }

// =====================================================================
// Weight pre-pack: OIHW f32 -> f16 B-fragment order, K padded to 32,
// N padded to 16*NT.  Layout: [kc][nt][lane][16 halves], where
//   k = kc*32 + (lane>=16 ? 16 : 0) + h,  n = nt*16 + (lane&15)
// =====================================================================
__global__ void pack_weights(const float* __restrict__ w, _Float16* __restrict__ bpk,
                             int CIN, int COUT, int K, int NT, int total) {
    int idx = blockIdx.x * blockDim.x + threadIdx.x;
    if (idx >= total) return;
    int h    = idx & 15;
    int lane = (idx >> 4) & 31;
    int nt   = (idx >> 9) % NT;
    int kc   = idx / (512 * NT);
    int k = kc * 32 + ((lane >> 4) << 4) + h;
    int n = nt * 16 + (lane & 15);
    float val = 0.f;
    if (k < K && n < COUT) {
        int c = k / 25, r = k - c * 25;
        val = w[(size_t)(n * CIN + c) * 25 + r];
    }
    bpk[idx] = (_Float16)val;
}

// =====================================================================
// Implicit-GEMM conv (5x5, stride 2, pad 2) via v_wmma_f32_16x16x32_f16.
// grid = (images, pixel-groups of 64); block = 32*NT threads.
// Each wave: one 16-channel N tile, 4 accumulators covering 64 pixels
// (B fragment loaded once per K chunk, reused by 4 WMMAs).
// A chunk (64 px x 32 K) staged in 4KB of LDS per iteration (im2col gather).
// =====================================================================
template <int CIN, int COUT, int HIN, int WIN, int HOUT, int WOUT, int NT,
          bool F32OUT, typename InT>
__global__ void conv_wmma(const InT* __restrict__ in, const _Float16* __restrict__ bpk,
                          const float* __restrict__ bias,
                          _Float16* __restrict__ outh, float* __restrict__ outf) {
    constexpr int K    = CIN * 25;
    constexpr int KC   = (K + 31) / 32;
    constexpr int NPIX = HOUT * WOUT;

    __shared__ __align__(16) _Float16 Asm[64 * 32];

    const int tid  = threadIdx.x;
    const int lane = tid & 31;
    const int wv   = tid >> 5;           // N-tile index
    const int img  = blockIdx.x;
    const int p0   = blockIdx.y * 64;    // first output pixel of this block

    v8f acc[4] = {};
    const int mA   = lane & 15;
    const int koff = (lane >> 4) * 16;

    for (int kc = 0; kc < KC; ++kc) {
        // ---- stage 64x32 im2col chunk (shift/mask indexing only) ----
        for (int idx = tid; idx < 64 * 32; idx += 32 * NT) {
            int m  = idx >> 5;
            int kk = idx & 31;
            int k  = kc * 32 + kk;
            int p  = p0 + m;
            float val = 0.f;
            if (k < K && p < NPIX) {
                int oy = p / WOUT, ox = p - oy * WOUT;
                int c  = k / 25, r = k - c * 25;
                int ky = r / 5,  kx = r - ky * 5;
                int iy = oy * 2 - 2 + ky, ix = ox * 2 - 2 + kx;
                if (iy >= 0 && iy < HIN && ix >= 0 && ix < WIN)
                    val = (float)in[((size_t)img * CIN + c) * (size_t)(HIN * WIN) + (size_t)iy * WIN + ix];
            }
            Asm[idx] = (_Float16)val;
        }
        __syncthreads();

        // ---- one B fragment, four WMMAs (4 M sub-tiles) ----
        v16h Bv = *(const v16h*)&bpk[((size_t)(kc * NT + wv) * 32 + lane) * 16];
#pragma unroll
        for (int s = 0; s < 4; ++s) {
            union { v16h v; v8h h[2]; } A;
            A.h[0] = *(const v8h*)&Asm[(s * 16 + mA) * 32 + koff];
            A.h[1] = *(const v8h*)&Asm[(s * 16 + mA) * 32 + koff + 8];
            acc[s] = __builtin_amdgcn_wmma_f32_16x16x32_f16(
                false, A.v, false, Bv, (short)0, acc[s], false, false);
        }
        __syncthreads();   // WAR: LDS re-staged next chunk
    }

    // ---- bias + ReLU + store (C/D layout: VGPR r -> rows mb+r, col lane&15) ----
    const int n  = lane & 15;
    const int mb = (lane >> 4) * 8;
    const int co = wv * 16 + n;
    if (co < COUT) {
        const float bco = bias[co];
#pragma unroll
        for (int s = 0; s < 4; ++s) {
#pragma unroll
            for (int r = 0; r < 8; ++r) {
                int p = p0 + s * 16 + mb + r;
                if (p < NPIX) {
                    float v = acc[s][r] + bco;
                    v = v > 0.f ? v : 0.f;
                    size_t off = ((size_t)img * COUT + co) * (size_t)NPIX + p;
                    if constexpr (F32OUT) outf[off] = v;
                    else                  outh[off] = (_Float16)v;
                }
            }
        }
    }
}

// =====================================================================
// Head: feats[n, f*8+k] = sum_p act3[n,f,p] * head_w[f,p,k] + head_b[f,k]
// grid = BT*NF blocks, 128 threads
// =====================================================================
__global__ void head_kernel(const float* __restrict__ act3, const float* __restrict__ hw,
                            const float* __restrict__ hb, float* __restrict__ feats) {
    const int n = blockIdx.x >> 3, f = blockIdx.x & 7;
    const int tid = threadIdx.x;
    float part[8] = {0, 0, 0, 0, 0, 0, 0, 0};
    const float* x = act3 + ((size_t)n * 8 + f) * 800;
    for (int p = tid; p < 800; p += 128) {
        float xv = x[p];
        const float* wr = hw + ((size_t)f * 800 + p) * 8;
#pragma unroll
        for (int k = 0; k < 8; ++k) part[k] += xv * wr[k];
    }
    __shared__ float red[128][8];
#pragma unroll
    for (int k = 0; k < 8; ++k) red[tid][k] = part[k];
    __syncthreads();
    for (int s = 64; s > 0; s >>= 1) {
        if (tid < s)
#pragma unroll
            for (int k = 0; k < 8; ++k) red[tid][k] += red[tid + s][k];
        __syncthreads();
    }
    if (tid < 8) feats[(size_t)n * 64 + f * 8 + tid] = red[0][tid] + hb[f * 8 + tid];
}

// ---- global std (ddof=1) over 8192 feats; one block of 256 ----
__global__ void std_kernel(const float* __restrict__ feats, float* __restrict__ stdv) {
    const int tid = threadIdx.x;
    float s = 0.f, s2 = 0.f;
    for (int i = tid; i < kBT * 64; i += 256) {
        float x = feats[i];
        s += x; s2 += x * x;
    }
    __shared__ float rs[256], rq[256];
    rs[tid] = s; rq[tid] = s2;
    __syncthreads();
    for (int o = 128; o > 0; o >>= 1) {
        if (tid < o) { rs[tid] += rs[tid + o]; rq[tid] += rq[tid + o]; }
        __syncthreads();
    }
    if (tid == 0) {
        const float N = (float)(kBT * 64);
        float var = (rq[0] - rs[0] * rs[0] / N) / (N - 1.0f);
        stdv[0] = sqrtf(fmaxf(var, 0.f)) + 1e-5f;
    }
}

// ---- speed MLP + normalize + input affine -> xs[t][b][72] ----
__global__ void xs_kernel(const float* __restrict__ feats, const float* __restrict__ speed,
                          const float* __restrict__ w1, const float* __restrict__ b1,
                          const float* __restrict__ w2, const float* __restrict__ b2,
                          const float* __restrict__ inw, const float* __restrict__ inb,
                          const float* __restrict__ stdv, float* __restrict__ xs) {
    const int nIdx = blockIdx.x;            // b*T + t
    const int b = nIdx >> 5, t = nIdx & 31;
    const int tid = threadIdx.x;
    __shared__ float hid[16], emb[8];
    const float s = speed[nIdx];
    if (tid < 16) { float h = s * w1[tid] + b1[tid]; hid[tid] = h > 0.f ? h : 0.f; }
    __syncthreads();
    if (tid < 8) {
        float a = b2[tid];
#pragma unroll
        for (int j = 0; j < 16; ++j) a += hid[j] * w2[j * 8 + tid];
        emb[tid] = tanhf(a);
    }
    __syncthreads();
    if (tid < kS) {
        float inv = 1.0f / stdv[0];
        float val = (tid < 64) ? feats[(size_t)nIdx * 64 + tid] * inv : emb[tid - 64];
        xs[((size_t)t * kB + b) * kS + tid] = val * inw[tid] + inb[tid];
    }
}

// =====================================================================
// LTC scan: one persistent block, 256 threads = (b,u); v shared via LDS.
// =====================================================================
__global__ void ltc_scan(const float* __restrict__ xs,
                         const float* __restrict__ gleak, const float* __restrict__ vleak,
                         const float* __restrict__ cm,
                         const float* __restrict__ sigma, const float* __restrict__ mu,
                         const float* __restrict__ wsyn,  const float* __restrict__ erev,
                         const float* __restrict__ ssig,  const float* __restrict__ smu,
                         const float* __restrict__ sw,    const float* __restrict__ serev,
                         const float* __restrict__ ow,    const float* __restrict__ ob,
                         float* __restrict__ out) {
    const int tid = threadIdx.x;
    const int b = tid >> 6, u = tid & 63;
    __shared__ float vsh[kB][kU];
    float v = 0.f;
    vsh[b][u] = 0.f;
    const float cmt  = cm[u] * 6.0f;
    const float gl   = gleak[u];
    const float glvl = gl * vleak[u];
    float* readout = out;            // (B,T,1): 128
    float* vlast   = out + 128;      // (B,U):   256
    float* allh    = out + 384;      // (B,T,U): 8192
    __syncthreads();
    for (int t = 0; t < kT; ++t) {
        const float* xr = xs + ((size_t)t * kB + b) * kS;
        float wns = 0.f, wds = 0.f;
        for (int j = 0; j < kS; ++j) {
            float a = sw[j * kU + u] * sigm_((xr[j] - smu[j * kU + u]) * ssig[j * kU + u]);
            wns += a * serev[j * kU + u];
            wds += a;
        }
        for (int it = 0; it < 6; ++it) {
            float num = wns, den = wds;
            for (int j = 0; j < kU; ++j) {
                float a = wsyn[j * kU + u] * sigm_((vsh[b][j] - mu[j * kU + u]) * sigma[j * kU + u]);
                num += a * erev[j * kU + u];
                den += a;
            }
            v = (cmt * v + glvl + num) / (cmt + gl + den + kEPS);
            __syncthreads();
            vsh[b][u] = v;
            __syncthreads();
        }
        allh[((size_t)b * kT + t) * kU + u] = v;
        if (u == 0) readout[b * kT + t] = v * ow[0] + ob[0];
    }
    vlast[b * kU + u] = v;
}

// =====================================================================
extern "C" void kernel_launch(void* const* d_in, const int* in_sizes, int n_in,
                              void* d_out, int out_size, void* d_ws, size_t ws_size,
                              hipStream_t stream) {
    const float* input   = (const float*)d_in[0];
    const float* speed   = (const float*)d_in[1];
    const float* c1w = (const float*)d_in[2];  const float* c1b = (const float*)d_in[3];
    const float* c2w = (const float*)d_in[4];  const float* c2b = (const float*)d_in[5];
    const float* c3w = (const float*)d_in[6];  const float* c3b = (const float*)d_in[7];
    const float* hw  = (const float*)d_in[8];  const float* hb  = (const float*)d_in[9];
    const float* sw1 = (const float*)d_in[10]; const float* sb1 = (const float*)d_in[11];
    const float* sw2 = (const float*)d_in[12]; const float* sb2 = (const float*)d_in[13];
    const float* inw = (const float*)d_in[14]; const float* inb = (const float*)d_in[15];
    const float* gleak = (const float*)d_in[16]; const float* vleak = (const float*)d_in[17];
    const float* cm    = (const float*)d_in[18];
    const float* sigma = (const float*)d_in[19]; const float* mu   = (const float*)d_in[20];
    const float* wsyn  = (const float*)d_in[21]; const float* erev = (const float*)d_in[22];
    const float* ssig  = (const float*)d_in[23]; const float* smu  = (const float*)d_in[24];
    const float* swn   = (const float*)d_in[25]; const float* serv = (const float*)d_in[26];
    const float* ow    = (const float*)d_in[27]; const float* ob   = (const float*)d_in[28];
    float* out = (float*)d_out;

    // ---- workspace bump allocator (256B aligned) ----
    char* ws = (char*)d_ws;
    size_t off = 0;
    auto alloc = [&](size_t bytes) { char* p = ws + off; off = (off + bytes + 255) & ~(size_t)255; return p; };
    _Float16* bpk1 = (_Float16*)alloc((size_t)3  * 1 * 512 * 2);
    _Float16* bpk2 = (_Float16*)alloc((size_t)13 * 2 * 512 * 2);
    _Float16* bpk3 = (_Float16*)alloc((size_t)25 * 1 * 512 * 2);
    _Float16* act1 = (_Float16*)alloc((size_t)kBT * 16 * 80 * 160 * 2);
    _Float16* act2 = (_Float16*)alloc((size_t)kBT * 32 * 40 * 80 * 2);
    float*    act3 = (float*)   alloc((size_t)kBT * 8 * 800 * 4);
    float*    feats= (float*)   alloc((size_t)kBT * 64 * 4);
    float*    stdv = (float*)   alloc(256);
    float*    xs   = (float*)   alloc((size_t)kT * kB * kS * 4);

    // ---- weight pre-pack into WMMA B-fragment order ----
    { int tot = 3 * 1 * 512;  pack_weights<<<(tot + 255) / 256, 256, 0, stream>>>(c1w, bpk1, 3,  16, 75,  1, tot); }
    { int tot = 13 * 2 * 512; pack_weights<<<(tot + 255) / 256, 256, 0, stream>>>(c2w, bpk2, 16, 32, 400, 2, tot); }
    { int tot = 25 * 1 * 512; pack_weights<<<(tot + 255) / 256, 256, 0, stream>>>(c3w, bpk3, 32, 8,  800, 1, tot); }

    // ---- conv stack (implicit-GEMM WMMA, 64-pixel M blocks) ----
    conv_wmma<3, 16, 160, 320, 80, 160, 1, false, float>
        <<<dim3(kBT, 200), 32, 0, stream>>>(input, bpk1, c1b, act1, nullptr);   // 12800/64
    conv_wmma<16, 32, 80, 160, 40, 80, 2, false, _Float16>
        <<<dim3(kBT, 50), 64, 0, stream>>>(act1, bpk2, c2b, act2, nullptr);     // 3200/64
    conv_wmma<32, 8, 40, 80, 20, 40, 1, true, _Float16>
        <<<dim3(kBT, 13), 32, 0, stream>>>(act2, bpk3, c3b, nullptr, act3);     // ceil(800/64)

    // ---- head, normalize, speed-MLP, scan ----
    head_kernel<<<kBT * kNF, 128, 0, stream>>>(act3, hw, hb, feats);
    std_kernel<<<1, 256, 0, stream>>>(feats, stdv);
    xs_kernel<<<kBT, 128, 0, stream>>>(feats, speed, sw1, sb1, sw2, sb2, inw, inb, stdv, xs);
    ltc_scan<<<1, 256, 0, stream>>>(xs, gleak, vleak, cm, sigma, mu, wsyn, erev,
                                    ssig, smu, swn, serv, ow, ob, out);
}
